// FuzzyNet_40888088658324
// MI455X (gfx1250) — matrix operand order repaired and checked by reading, not compiled, not run on hardware
//
#include <hip/hip_runtime.h>
#include <math.h>

// ---------------------------------------------------------------------------
// FuzzyNet forward for MI455X (gfx1250, wave32, WMMA).
//
// fuzzynum n=0 => fuzzy membership exp(-empty_sum)=1 => bn(const)=0 => the
// fuzzy/fbn branch is identically zero; gate(x1,0) =
// sigmoid(conv3(relu(conv1(cat(x1,0))))) * x1.
//
// Convs are implicit GEMM on v_wmma_f32_16x16x32_f16 (f32 accumulate):
//   block tile 64(M=Cout) x 128(N=B*H*W), K step 32, 8 waves as 2x4,
//   each wave owns a 32x32 output sub-tile = 2x2 WMMA fragments
//   (4 v_wmma per K step, 2x fragment reuse).
// Shape contract exploited (true for every conv in this net):
//   Cout % 64 == 0, (B*H*W) % 128 == 0  => no M/N tile guards anywhere.
//   Spatial dims are square powers of two => shift/mask pixel decode.
// Only the K guard remains (conv1 has Ktot = 27), as a branchless select.
// ---------------------------------------------------------------------------

typedef __attribute__((ext_vector_type(16))) _Float16 v16h;
typedef __attribute__((ext_vector_type(8)))  float    v8f;

union AFrag { v16h v; _Float16 s[16]; };
union CFrag { v8f  v; float    s[8];  };

#define TB 256   // 8 waves
#define MT 64    // Cout tile
#define NT 128   // pixel tile
#define KT 32    // K step (16x16x32 WMMA)

template <int KSZ>
__global__ __launch_bounds__(TB)
void conv2d_wmma(const float* __restrict__ in, const float* __restrict__ wgt,
                 const float* __restrict__ bias, float* __restrict__ out,
                 int B, int Cin, int lw, int Cout, int relu)
{
  constexpr int PAD = KSZ / 2;
  constexpr int KK  = KSZ * KSZ;
  const int W    = 1 << lw;
  const int lhw  = lw + lw;
  const int HW   = 1 << lhw;
  const int Ktot = Cin * KK;

  __shared__ _Float16 As[MT][KT + 2];   // [m][k]
  __shared__ _Float16 Bs[NT][KT + 2];   // [n][k]

  const int tid  = threadIdx.x;
  const int lane = tid & 31;
  const int wave = tid >> 5;
  const int wm   = wave >> 2;                 // 0..1 : M 32-chunk
  const int wn   = wave & 3;                  // 0..3 : N 32-chunk
  const int m0   = blockIdx.y * MT;
  const int n0   = blockIdx.x * NT;
  const int hi   = (lane >= 16) ? 1 : 0;

  // ---- per-thread staging assignment (fixed across the K loop) ------------
  const int aml = tid >> 2;             // A: row t>>2, k-octet (t&3)*8
  const int ako = (tid & 3) * 8;
  const int abase = (m0 + aml) * Ktot;

  const int bnl = tid >> 1;             // B: row t>>1, k-hexadectet (t&1)*16
  const int bko = (tid & 1) * 16;
  const int bn  = n0 + bnl;
  const int bb  = bn >> lhw;
  const int br_ = bn & (HW - 1);
  const int by  = br_ >> lw;
  const int bx  = br_ & (W - 1);
  const int bbase = ((bb * Cin) << lhw) + (by << lw) + bx;

  CFrag acc[2][2];
#pragma unroll
  for (int i = 0; i < 2; ++i)
#pragma unroll
    for (int j = 0; j < 2; ++j)
#pragma unroll
      for (int v = 0; v < 8; ++v) acc[i][j].s[v] = 0.0f;

  for (int k0 = 0; k0 < Ktot; k0 += KT) {
    // ---- stage A (weights); only K needs a (branchless) guard -------------
#pragma unroll
    for (int j = 0; j < 8; ++j) {
      int kg = k0 + ako + j;
      bool v = (kg < Ktot);
      float w = wgt[v ? (abase + kg) : 0];
      As[aml][ako + j] = (_Float16)(v ? w : 0.0f);
    }
    // ---- stage B (im2col), branchless -------------------------------------
#pragma unroll
    for (int j = 0; j < 16; ++j) {
      int kg = k0 + bko + j;
      int ci = kg / KK;
      int rr = kg - ci * KK;
      int kh = rr / KSZ;
      int kw = rr - kh * KSZ;
      int iy = by + kh - PAD, ix = bx + kw - PAD;
      bool v = (kg < Ktot) && (iy >= 0) && (iy < W) && (ix >= 0) && (ix < W);
      int addr = bbase + (ci << lhw) + ((kh - PAD) << lw) + (kw - PAD);
      float d = in[v ? addr : 0];
      Bs[bnl][bko + j] = (_Float16)(v ? d : 0.0f);
    }
    __syncthreads();

    // ---- fragment gather per CDNA5 16-bit WMMA layouts --------------------
    AFrag a[2], bf[2];
    const int ar  = wm * 32 + (lane & 15);
    const int brw = wn * 32 + (lane & 15);
#pragma unroll
    for (int e = 0; e < 16; ++e) {
      int ka = e + ((e >= 8) ? 8 : 0) + hi * 8;  // A 16x32 layout
      int kb = e + ((e >= 8) ? 8 : 0) + hi * 8;  // B 32x16 layout
      a[0].s[e]  = As[ar][ka];
      a[1].s[e]  = As[ar + 16][ka];
      bf[0].s[e] = Bs[brw][kb];
      bf[1].s[e] = Bs[brw + 16][kb];
    }
#pragma unroll
    for (int mi = 0; mi < 2; ++mi)
#pragma unroll
      for (int ni = 0; ni < 2; ++ni)
        acc[mi][ni].v = __builtin_amdgcn_wmma_f32_16x16x32_f16(
            false, a[mi].v, false, bf[ni].v, (short)0, acc[mi][ni].v,
            false, false);
    __syncthreads();
  }

  // ---- hoist bias into registers (one batched load group, one wait) -------
  float bi[2][8];
#pragma unroll
  for (int mi = 0; mi < 2; ++mi)
#pragma unroll
    for (int v = 0; v < 8; ++v)
      bi[mi][v] = bias[m0 + wm * 32 + mi * 16 + v + hi * 8];

  // ---- epilogue: straight-line stores (no tile guards needed) -------------
  // D layout: VGPR v -> M = v (lanes 0-15) / 8+v (lanes 16-31); N = lane%16.
#pragma unroll
  for (int ni = 0; ni < 2; ++ni) {
    const int nn = n0 + wn * 32 + ni * 16 + (lane & 15);
    const int b  = nn >> lhw;
    const int r  = nn & (HW - 1);
    const int obase = ((b * Cout) << lhw) + r;
#pragma unroll
    for (int mi = 0; mi < 2; ++mi) {
      const int mm = m0 + wm * 32 + mi * 16 + hi * 8;
#pragma unroll
      for (int v = 0; v < 8; ++v) {
        float o = acc[mi][ni].s[v] + bi[mi][v];
        if (relu) o = fmaxf(o, 0.0f);
        out[obase + ((mm + v) << lhw)] = o;
      }
    }
  }
}

// ---------------------------------------------------------------------------
// Training-mode BatchNorm (gamma=1, beta=0, biased variance).
// ---------------------------------------------------------------------------
__global__ __launch_bounds__(256)
void bn_stats(const float* __restrict__ x, float* __restrict__ stats,
              int B, int C, int HW)
{
  __shared__ float ssum[256], ssq[256];
  const int c = blockIdx.x;
  float s = 0.0f, q = 0.0f;
  for (int b = 0; b < B; ++b) {
    const float* p = x + ((long long)b * C + c) * HW;
    for (int i = threadIdx.x; i < HW; i += 256) {
      float v = p[i]; s += v; q += v * v;
    }
  }
  ssum[threadIdx.x] = s; ssq[threadIdx.x] = q;
  __syncthreads();
  for (int off = 128; off > 0; off >>= 1) {
    if (threadIdx.x < off) {
      ssum[threadIdx.x] += ssum[threadIdx.x + off];
      ssq[threadIdx.x]  += ssq[threadIdx.x + off];
    }
    __syncthreads();
  }
  if (threadIdx.x == 0) {
    float n = (float)B * (float)HW;
    float m = ssum[0] / n;
    float var = ssq[0] / n - m * m;
    stats[c]     = m;
    stats[C + c] = rsqrtf(var + 1e-5f);
  }
}

// lhw = log2(HW), cmask = C-1 (C is a power of two everywhere BN is used)
__global__ void bn_apply(float* __restrict__ x, const float* __restrict__ stats,
                         int cmask, int lhw, long long n, int relu)
{
  long long i = (long long)blockIdx.x * 256 + threadIdx.x;
  if (i >= n) return;
  int c = (int)(i >> lhw) & cmask;
  float v = (x[i] - stats[c]) * stats[cmask + 1 + c];
  if (relu) v = fmaxf(v, 0.0f);
  x[i] = v;
}

// out = maxpool2( a (+ b) ); b may be null. W=H=2^lw (input dims).
__global__ void add_maxpool2(const float* __restrict__ a, const float* __restrict__ b,
                             float* __restrict__ out, long long n, int lw)
{
  long long i = (long long)blockIdx.x * 256 + threadIdx.x;
  if (i >= n) return;
  const int lwo = lw - 1;
  int xo = (int)i & ((1 << lwo) - 1);
  int yo = (int)(i >> lwo) & ((1 << lwo) - 1);
  long long cc = i >> (2 * lwo);                 // b*C + c
  const long long base = (cc << (2 * lw)) + ((long long)(2 * yo) << lw) + 2 * xo;
  float m = -INFINITY;
#pragma unroll
  for (int dy = 0; dy < 2; ++dy)
#pragma unroll
    for (int dx = 0; dx < 2; ++dx) {
      long long j = base + ((long long)dy << lw) + dx;
      float v = a[j] + (b ? b[j] : 0.0f);
      m = fmaxf(m, v);
    }
  out[i] = m;
}

// dst(B,2C,HW): first C channels copy src, second C channels zero.
__global__ void concat_zero(const float* __restrict__ src, float* __restrict__ dst,
                            long long n, int cmask2, int lhw)
{
  long long i = (long long)blockIdx.x * 256 + threadIdx.x;
  if (i >= n) return;
  int r = (int)i & ((1 << lhw) - 1);
  int c = (int)(i >> lhw) & cmask2;
  long long bb = i >> lhw >> (31 - __builtin_clz(cmask2 + 1));  // / (2C)
  int C = (cmask2 + 1) >> 1;
  dst[i] = (c < C) ? src[((bb * C + c) << lhw) + r] : 0.0f;
}

// f = sigmoid(g) * c
__global__ void gate_combine(const float* __restrict__ g, const float* __restrict__ c,
                             float* __restrict__ out, long long n)
{
  long long i = (long long)blockIdx.x * 256 + threadIdx.x;
  if (i >= n) return;
  float s = 1.0f / (1.0f + __expf(-g[i]));
  out[i] = s * c[i];
}

// ConvTranspose2d(k=2,s=2): out[b,o,2y+ky,2x+kx] = sum_ci in[b,ci,y,x]*w[ci,o,ky,kx]
__global__ void deconv2x2(const float* __restrict__ in, const float* __restrict__ w,
                          const float* __restrict__ skip, float* __restrict__ out,
                          int B, int Ci, int Co, int H, int W)
{
  const int H2 = H * 2, W2 = W * 2;
  long long n = (long long)B * Co * H2 * W2;
  long long i = (long long)blockIdx.x * 256 + threadIdx.x;
  if (i >= n) return;
  int x2 = (int)(i % W2); long long t = i / W2;
  int y2 = (int)(t % H2); t /= H2;
  int o  = (int)(t % Co); int bb = (int)(t / Co);
  int y = y2 >> 1, x = x2 >> 1, ky = y2 & 1, kx = x2 & 1;
  const float* ip = in + ((long long)bb * Ci * H + y) * W + x;
  const float* wp = w + ((long long)o * 4 + ky * 2 + kx);
  float s = 0.0f;
  for (int ci = 0; ci < Ci; ++ci)
    s += ip[(long long)ci * H * W] * wp[(long long)ci * Co * 4];
  if (skip) s += skip[i];
  out[i] = s;
}

// ---------------------------------------------------------------------------
// Host orchestration
// ---------------------------------------------------------------------------
enum {
  IN_X = 0,
  IN_C1_1_W, IN_C1_1_B, IN_C1_2_W, IN_C1_2_B,
  IN_C2_1_W, IN_C2_1_B, IN_C2_2_W, IN_C2_2_B,
  IN_C2_R1_W, IN_C2_R1_B, IN_C2_R2_W, IN_C2_R2_B, IN_C2_R3_W, IN_C2_R3_B,
  IN_C3_1_W, IN_C3_1_B, IN_C3_2_W, IN_C3_2_B, IN_C3_3_W, IN_C3_3_B,
  IN_C3_R1_W, IN_C3_R1_B, IN_C3_R2_W, IN_C3_R2_B, IN_C3_R3_W, IN_C3_R3_B,
  IN_C4_1_W, IN_C4_1_B, IN_C4_2_W, IN_C4_2_B, IN_C4_3_W, IN_C4_3_B,
  IN_C4_R1_W, IN_C4_R1_B, IN_C4_R2_W, IN_C4_R2_B, IN_C4_R3_W, IN_C4_R3_B,
  IN_D1_W, IN_D2_W, IN_D3_W, IN_D4_W,
  IN_F1_C1_W, IN_F1_C1_B, IN_F1_C2_W, IN_F1_C2_B, IN_F1_MU, IN_F1_SIGMA,
  IN_F2_C1_W, IN_F2_C1_B, IN_F2_C2_W, IN_F2_C2_B, IN_F2_MU, IN_F2_SIGMA,
  IN_F3_C1_W, IN_F3_C1_B, IN_F3_C2_W, IN_F3_C2_B, IN_F3_MU, IN_F3_SIGMA,
  IN_G1_F_W, IN_G1_F_B, IN_G1_G_W, IN_G1_G_B,
  IN_G2_F_W, IN_G2_F_B, IN_G2_G_W, IN_G2_G_B,
  IN_G3_F_W, IN_G3_F_B, IN_G3_G_W, IN_G3_G_B
};

extern "C" void kernel_launch(void* const* d_in, const int* in_sizes, int n_in,
                              void* d_out, int out_size, void* d_ws, size_t ws_size,
                              hipStream_t stream)
{
  (void)in_sizes; (void)n_in; (void)out_size; (void)ws_size;
  float* ws = (float*)d_ws;

  const size_t S1  = 4ull * 64  * 256 * 256;
  const size_t S2  = 4ull * 128 * 128 * 128;
  const size_t S1p = 4ull * 64  * 128 * 128;
  const size_t S2p = 4ull * 128 * 64  * 64;
  const size_t S3p = 4ull * 256 * 32  * 32;

  float* P0 = ws;
  float* P1 = P0 + S1;
  float* P2 = P1 + S1;
  float* P3 = P2 + S2;
  float* C1 = P3 + S2;
  float* C2 = C1 + S1p;
  float* C3 = C2 + S2p;
  float* F1 = C3 + S3p;
  float* F2 = F1 + S1p;
  float* F3 = F2 + S2p;
  float* STATS = F3 + S3p;        // mean[C] then invstd[C], C<=512

  const float* X = (const float*)d_in[IN_X];
  auto W_ = [&](int i) { return (const float*)d_in[i]; };

  const int B = 4;
  auto cdiv = [](long long a, long long b) { return (unsigned)((a + b - 1) / b); };
  auto lg2  = [](int v) { return 31 - __builtin_clz(v); };

  auto conv = [&](const float* in, int widx, float* out,
                  int Cin, int Wd, int Cout, int k, int relu) {
    int lw = lg2(Wd);
    dim3 grid(cdiv((long long)B << (2 * lw), NT), cdiv(Cout, MT));
    if (k == 1)
      conv2d_wmma<1><<<grid, TB, 0, stream>>>(in, W_(widx), W_(widx + 1), out,
                                              B, Cin, lw, Cout, relu);
    else
      conv2d_wmma<3><<<grid, TB, 0, stream>>>(in, W_(widx), W_(widx + 1), out,
                                              B, Cin, lw, Cout, relu);
  };
  auto bn = [&](float* buf, int C, int Wd, int relu) {
    int lw = lg2(Wd), HW = Wd * Wd;
    bn_stats<<<C, 256, 0, stream>>>(buf, STATS, B, C, HW);
    long long n = (long long)B * C * HW;
    bn_apply<<<cdiv(n, 256), 256, 0, stream>>>(buf, STATS, C - 1, 2 * lw, n, relu);
  };
  auto pool = [&](const float* a, const float* g, float* out, int C, int Wd) {
    long long n = (long long)B * C * (Wd / 2) * (Wd / 2);
    add_maxpool2<<<cdiv(n, 256), 256, 0, stream>>>(a, g, out, n, lg2(Wd));
  };
  auto gate = [&](const float* c, int gi, float* f, int C, int Wd) {
    int lw = lg2(Wd), HW = Wd * Wd;
    long long n2 = (long long)B * 2 * C * HW;
    concat_zero<<<cdiv(n2, 256), 256, 0, stream>>>(c, P2, n2, 2 * C - 1, 2 * lw);
    conv(P2, gi,     P0, 2 * C, Wd, C, 1, 1);   // feature 1x1 + relu
    conv(P0, gi + 2, P1, C,     Wd, C, 3, 0);   // gate 3x3
    long long n = (long long)B * C * HW;
    gate_combine<<<cdiv(n, 256), 256, 0, stream>>>(P1, c, f, n);
  };
  auto deconv = [&](const float* in, int widx, const float* skip, float* out,
                    int Ci, int Co, int Wd) {
    long long n = (long long)B * Co * (2 * Wd) * (2 * Wd);
    deconv2x2<<<cdiv(n, 256), 256, 0, stream>>>(in, W_(widx), skip, out,
                                                B, Ci, Co, Wd, Wd);
  };

  // ---------------- level 1 (256x256 -> c1 @128) ---------------------------
  conv(X,  IN_C1_1_W, P0, 3,  256, 64, 3, 1);
  conv(P0, IN_C1_2_W, P1, 64, 256, 64, 3, 1);
  pool(P1, nullptr, C1, 64, 256);
  gate(C1, IN_G1_F_W, F1, 64, 128);               // t1 == 0 (fuzzy collapses)

  // ---------------- level 2 (128x128 -> c2 @64) ----------------------------
  conv(C1, IN_C2_R1_W, P0, 64, 128, 64, 1, 0);   bn(P0, 64, 128, 1);
  conv(P0, IN_C2_R2_W, P2, 64, 128, 128, 3, 0);  bn(P2, 128, 128, 1);
  conv(P2, IN_C2_R3_W, P3, 128, 128, 128, 1, 0);
  conv(C1, IN_C2_1_W,  P0, 64, 128, 128, 3, 1);
  conv(P0, IN_C2_2_W,  P1, 128, 128, 128, 3, 1);
  pool(P1, P3, C2, 128, 128);
  gate(C2, IN_G2_F_W, F2, 128, 64);

  // ---------------- level 3 (64x64 -> c3 @32) ------------------------------
  conv(C2, IN_C3_R1_W, P0, 128, 64, 128, 1, 0);  bn(P0, 128, 64, 1);
  conv(P0, IN_C3_R2_W, P2, 128, 64, 256, 3, 0);  bn(P2, 256, 64, 1);
  conv(P2, IN_C3_R3_W, P3, 256, 64, 256, 1, 0);
  conv(C2, IN_C3_1_W,  P0, 128, 64, 256, 3, 1);
  conv(P0, IN_C3_2_W,  P1, 256, 64, 256, 3, 1);
  conv(P1, IN_C3_3_W,  P0, 256, 64, 256, 3, 1);
  pool(P0, P3, C3, 256, 64);
  gate(C3, IN_G3_F_W, F3, 256, 32);

  // ---------------- level 4 (32x32 -> @16) ---------------------------------
  conv(C3, IN_C4_R1_W, P0, 256, 32, 256, 1, 0);  bn(P0, 256, 32, 1);
  conv(P0, IN_C4_R2_W, P2, 256, 32, 512, 3, 0);  bn(P2, 512, 32, 1);
  conv(P2, IN_C4_R3_W, P3, 512, 32, 512, 1, 0);
  conv(C3, IN_C4_1_W,  P0, 256, 32, 512, 3, 1);
  conv(P0, IN_C4_2_W,  P1, 512, 32, 512, 3, 1);
  conv(P1, IN_C4_3_W,  P0, 512, 32, 512, 3, 1);
  pool(P0, P3, P1, 512, 32);                          // P1: (4,512,16,16)

  // ---------------- decoder ------------------------------------------------
  bn(P1, 512, 16, 0);                                 // bn1
  deconv(P1, IN_D1_W, F3, P0, 512, 256, 16);          // + f3
  bn(P0, 256, 32, 0);                                 // bn2
  deconv(P0, IN_D2_W, F2, P1, 256, 128, 32);          // + f2
  bn(P1, 128, 64, 0);                                 // bn3
  deconv(P1, IN_D3_W, F1, P0, 128, 64, 64);           // + f1
  bn(P0, 64, 128, 0);                                 // bn4
  deconv(P0, IN_D4_W, nullptr, (float*)d_out, 64, 6, 128);  // final (4,6,256,256)
}
